// GRUTimeSeriesBinaryClassifier_74062416053155
// MI455X (gfx1250) — matrix operand order repaired and checked by reading, not compiled, not run on hardware
//
#include <hip/hip_runtime.h>
#include <cstdint>
#include <cstddef>

// ---- problem dims (fixed by the reference) ----
#define GB    64              // batch
#define GT    512             // time steps
#define GF    2048            // input features
#define GH1   128             // layer-1 hidden
#define GH2   64              // layer-2 hidden
#define XGC   384             // 3*GH1 gate columns
#define XG_SLAB (GB * XGC)    // floats per timestep slab = 24576 (96 KB)

typedef __attribute__((ext_vector_type(16))) __bf16       v16bf;
typedef __attribute__((ext_vector_type(8)))  float        v8f;
typedef __attribute__((ext_vector_type(4)))  unsigned int u32x4;
typedef __attribute__((ext_vector_type(8)))  int          i32x8;
typedef __attribute__((ext_vector_type(4)))  int          i32x4;

#if defined(__HIP_DEVICE_COMPILE__)

__device__ __forceinline__ v8f wmma_bf16(v16bf a, v16bf b, v8f c) {
  // D = A(16x32 bf16) * B(32x16 bf16) + C(16x16 f32)
  return __builtin_amdgcn_wmma_f32_16x16x32_bf16(false, a, false, b,
                                                 (short)0, c, false, false);
}

// A fragment (16x32, MxK): lanes 0-15 hold row M=lane, K = k0..k0+15;
// lanes 16-31 hold row M=lane-16, K = k0+16..k0+31.
// Caller passes p = row_base + k0 + (lane>=16 ? 16 : 0); we read p[0..15].
__device__ __forceinline__ v16bf load_a_f32(const float* p) {
  v16bf a;
#pragma unroll
  for (int i = 0; i < 8; ++i) a[i] = (__bf16)p[i];
#pragma unroll
  for (int i = 0; i < 8; ++i) a[8 + i] = (__bf16)p[8 + i];
  return a;
}

// B fragment (32x16, KxN) for B = Wt (i.e. B[k][n] = W[n][k], W row-major):
// lanes 0-15: col n = n0+lane, K = k0..k0+15; lanes 16-31: col n0+lane-16, K = k0+16..k0+31.
// Caller passes p = &W[n0 + lane%16][k0 + (lane>=16 ? 16 : 0)]; 16 contiguous floats.
__device__ __forceinline__ v16bf load_b_f32(const float* p) {
  v16bf b;
#pragma unroll
  for (int i = 0; i < 16; ++i) b[i] = (__bf16)p[i];
  return b;
}

__device__ __forceinline__ float sigmoid_f(float x) {
  return 1.0f / (1.0f + __expf(-x));
}

// Tensor Data Mover: async copy of `nelem` contiguous f32 (global -> LDS).
// D# per CDNA5 ISA ch.8: group0 = {count=1 | lds_addr | global_addr | type=2},
// group1 = {data_size=4B, tensor_dim0=nelem, tensor_dim1=1, tile_dim0=nelem,
//           tile_dim1=1, tensor_dim0_stride=nelem}.
__device__ __forceinline__ void tdm_load_slab(unsigned lds_off,
                                              const float* gptr,
                                              unsigned nelem /* fits 16 bits */) {
  unsigned long long ga = (unsigned long long)(size_t)gptr;
  u32x4 g0;
  g0[0] = 1u;                                     // count=1, user descriptor
  g0[1] = lds_off;                                // lds_addr [63:32]
  g0[2] = (unsigned)ga;                           // global_addr lo
  g0[3] = (unsigned)((ga >> 32) & 0x01FFFFFFull)  // global_addr [56:32]
        | (2u << 30);                             // type = 2 ("image")
  i32x8 g1;
  g1[0] = (2 << 16);                              // data_size = 2 -> 4 bytes
  g1[1] = (int)((nelem & 0xFFFFu) << 16);         // tensor_dim0[15:0] @ bits[63:48]
  g1[2] = (int)((1u << 16) | (nelem >> 16));      // tensor_dim0 hi, tensor_dim1=1
  g1[3] = (int)((nelem & 0xFFFFu) << 16);         // tile_dim0 @ bits[127:112]
  g1[4] = 1;                                      // tile_dim1 = 1, tile_dim2 = 0
  g1[5] = (int)nelem;                             // tensor_dim0_stride lo
  g1[6] = 0;                                      // stride hi / tensor_dim1_stride
  g1[7] = 0;
  i32x4 z4 = {0, 0, 0, 0};
#if __clang_major__ >= 23
  i32x8 z8 = {0, 0, 0, 0, 0, 0, 0, 0};
  __builtin_amdgcn_tensor_load_to_lds(g0, g1, z4, z4, z8, 0);
#else
  __builtin_amdgcn_tensor_load_to_lds(g0, g1, z4, z4, 0);
#endif
}

#endif // __HIP_DEVICE_COMPILE__

// ---------------------------------------------------------------------------
// Kernel 1: xg[t][b][0:384] = x[b,t,:] @ w_ih1^T + b_ih1   (bf16 WMMA, f32 acc)
// M = B*T = 32768 rows, N = 384, K = 2048.
// Each wave: one 16-row M-tile, all 24 N-tiles in registers (24 x v8f).
// ---------------------------------------------------------------------------
__global__ __launch_bounds__(256) void gru_l1_input_gemm(
    const float* __restrict__ x,      // [B,T,F] row-major -> [M, K]
    const float* __restrict__ w_ih1,  // [384, 2048]
    const float* __restrict__ b_ih1,  // [384]
    float* __restrict__ xg)           // [T][B][384]
{
#if defined(__HIP_DEVICE_COMPILE__)
  const int lane = threadIdx.x & 31;
  const int wave = threadIdx.x >> 5;
  const int hi   = lane >> 4;        // 0: lanes 0-15, 1: lanes 16-31
  const int lr   = lane & 15;
  const int hi8  = hi * 8;
  const int hi16 = hi * 16;

  const int m0 = (blockIdx.x * 8 + wave) * 16;   // 16-row tile base
  const float* arow = x + (size_t)(m0 + lr) * GF;

  v8f acc[24] = {};

#pragma unroll 1
  for (int k0 = 0; k0 < GF; k0 += 32) {
    v16bf a = load_a_f32(arow + k0 + hi16);
    __builtin_prefetch(arow + k0 + 512, 0, 1);   // stream-ahead on x
#pragma unroll
    for (int nt = 0; nt < 24; ++nt) {
      const float* bp = w_ih1 + (size_t)(nt * 16 + lr) * GF + k0 + hi16;
      acc[nt] = wmma_bf16(a, load_b_f32(bp), acc[nt]);
    }
  }

  // Store time-major with bias: row m = b*T + t  ->  xg[(m % T)][(m / T)][col]
#pragma unroll
  for (int nt = 0; nt < 24; ++nt) {
    const int col = nt * 16 + lr;
    const float bias = b_ih1[col];
#pragma unroll
    for (int v = 0; v < 8; ++v) {
      const int m = m0 + hi8 + v;                // C/D layout: M = v + 8*hi
      xg[(size_t)(m & (GT - 1)) * XG_SLAB + (size_t)(m >> 9) * XGC + col] =
          acc[nt][v] + bias;
    }
  }
#endif
}

// ---------------------------------------------------------------------------
// Kernel 2: persistent single-workgroup fused sequential part.
// 1024 threads = 32 waves on one WGP. LDS: h1 32KB + h2 16KB + xg slab 96KB.
// Per step t:
//   phase 1: hg1 = h1 @ w_hh1^T  (32 waves; wave (rt,jt) owns gates r/z/n of
//            its 16x16 tile -> 3 WMMA accumulators, K=128)
//            wave0 waits on the TDM slab for step t
//   phase 2: in-register gate math using xg slab from LDS; write h1
//            wave0 issues TDM prefetch of slab t+1
//   phase 3: layer-2 projections (16 waves, 6 accumulators each)
//   phase 4: layer-2 gate math; write h2
// Tail: FC head out[b] = h2[b,:] . fc_w + fc_b
// ---------------------------------------------------------------------------
__global__ __launch_bounds__(1024) void gru_seq_fused(
    const float* __restrict__ xg,     // [T][B][384] (includes b_ih1)
    const float* __restrict__ w_hh1,  // [384, 128]
    const float* __restrict__ b_hh1,  // [384]
    const float* __restrict__ w_ih2,  // [192, 128]
    const float* __restrict__ b_ih2,  // [192]
    const float* __restrict__ w_hh2,  // [192, 64]
    const float* __restrict__ b_hh2,  // [192]
    const float* __restrict__ fc_w,   // [64]
    const float* __restrict__ fc_b,   // [1]
    float* __restrict__ out)          // [64]
{
#if defined(__HIP_DEVICE_COMPILE__)
  __shared__ float h1[GB][GH1];          // 32 KB
  __shared__ float h2[GB][GH2];          // 16 KB
  __shared__ float xg_t[XG_SLAB];        // 96 KB, filled by TDM

  const int tid  = threadIdx.x;
  const int lane = tid & 31;
  const int wave = tid >> 5;
  const int hi   = lane >> 4;
  const int lr   = lane & 15;
  const int hi8  = hi * 8;
  const int hi16 = hi * 16;

  // zero initial hidden state
  for (int i = tid; i < GB * GH1; i += 1024) (&h1[0][0])[i] = 0.0f;
  for (int i = tid; i < GB * GH2; i += 1024) (&h2[0][0])[i] = 0.0f;
  __syncthreads();

  const unsigned lds_xg = (unsigned)(size_t)(void*)xg_t;
  if (wave == 0) tdm_load_slab(lds_xg, xg, XG_SLAB);   // prefetch slab t=0

  // layer-1 wave task: rt in 0..3 (rows), jt in 0..7 (hidden cols)
  const int rt  = wave >> 3;
  const int jt  = wave & 7;
  const int ar1 = rt * 16 + lr;                        // A row (h1)
  const float* w1r = w_hh1 + (size_t)(jt * 16 + lr) * GH1;
  const float* w1z = w1r + 128 * GH1;
  const float* w1n = w1r + 256 * GH1;
  const int j1 = jt * 16 + lr;
  const float bh1r = b_hh1[j1];
  const float bh1z = b_hh1[128 + j1];
  const float bh1n = b_hh1[256 + j1];

  // layer-2 wave task (waves 0..15): rt2 in 0..3, jt2 in 0..3
  const int rt2 = (wave & 15) >> 2;
  const int jt2 = wave & 3;
  const int ar2 = rt2 * 16 + lr;
  const float* w2ir = w_ih2 + (size_t)(jt2 * 16 + lr) * GH1;
  const float* w2iz = w2ir + 64 * GH1;
  const float* w2in = w2ir + 128 * GH1;
  const float* w2hr = w_hh2 + (size_t)(jt2 * 16 + lr) * GH2;
  const float* w2hz = w2hr + 64 * GH2;
  const float* w2hn = w2hr + 128 * GH2;
  const int j2 = jt2 * 16 + lr;

#pragma unroll 1
  for (int t = 0; t < GT; ++t) {
    // ---- phase 1: layer-1 recurrent GEMM (reads h1) ----
    v8f cr = {}, cz = {}, cn = {};
#pragma unroll
    for (int k0 = 0; k0 < GH1; k0 += 32) {
      v16bf a = load_a_f32(&h1[ar1][k0 + hi16]);
      const int kb = k0 + hi16;
      cr = wmma_bf16(a, load_b_f32(w1r + kb), cr);
      cz = wmma_bf16(a, load_b_f32(w1z + kb), cz);
      cn = wmma_bf16(a, load_b_f32(w1n + kb), cn);
    }
    if (wave == 0) __builtin_amdgcn_s_wait_tensorcnt(0);  // xg slab t ready
    __syncthreads();

    // ---- phase 2: layer-1 gates, update h1 (each lane owns its (b,j)) ----
    {
#pragma unroll
      for (int v = 0; v < 8; ++v) {
        const int b = rt * 16 + hi8 + v;
        const float xr = xg_t[b * XGC + j1];
        const float xz = xg_t[b * XGC + 128 + j1];
        const float xn = xg_t[b * XGC + 256 + j1];
        const float hp = h1[b][j1];
        const float r  = sigmoid_f(xr + cr[v] + bh1r);
        const float z  = sigmoid_f(xz + cz[v] + bh1z);
        const float n  = tanhf(xn + r * (cn[v] + bh1n));
        h1[b][j1] = (1.0f - z) * n + z * hp;
      }
    }
    __syncthreads();                       // h1 complete; xg_t reads done
    if (wave == 0 && t + 1 < GT)
      tdm_load_slab(lds_xg, xg + (size_t)(t + 1) * XG_SLAB, XG_SLAB);

    // ---- phase 3: layer-2 projections (16 waves) ----
    v8f xr2 = {}, xz2 = {}, xn2 = {}, hr2 = {}, hz2 = {}, hn2 = {};
    if (wave < 16) {
#pragma unroll
      for (int k0 = 0; k0 < GH1; k0 += 32) {          // h1 @ w_ih2^T
        v16bf a = load_a_f32(&h1[ar2][k0 + hi16]);
        const int kb = k0 + hi16;
        xr2 = wmma_bf16(a, load_b_f32(w2ir + kb), xr2);
        xz2 = wmma_bf16(a, load_b_f32(w2iz + kb), xz2);
        xn2 = wmma_bf16(a, load_b_f32(w2in + kb), xn2);
      }
#pragma unroll
      for (int k0 = 0; k0 < GH2; k0 += 32) {          // h2 @ w_hh2^T
        v16bf a = load_a_f32(&h2[ar2][k0 + hi16]);
        const int kb = k0 + hi16;
        hr2 = wmma_bf16(a, load_b_f32(w2hr + kb), hr2);
        hz2 = wmma_bf16(a, load_b_f32(w2hz + kb), hz2);
        hn2 = wmma_bf16(a, load_b_f32(w2hn + kb), hn2);
      }
    }
    __syncthreads();                       // all h2 reads done before update

    // ---- phase 4: layer-2 gates, update h2 ----
    if (wave < 16) {
      const float brz = b_ih2[j2] + b_hh2[j2];
      const float bzz = b_ih2[64 + j2] + b_hh2[64 + j2];
      const float bin = b_ih2[128 + j2];
      const float bhn = b_hh2[128 + j2];
#pragma unroll
      for (int v = 0; v < 8; ++v) {
        const int b = rt2 * 16 + hi8 + v;
        const float hp = h2[b][j2];
        const float r  = sigmoid_f(xr2[v] + hr2[v] + brz);
        const float z  = sigmoid_f(xz2[v] + hz2[v] + bzz);
        const float n  = tanhf(xn2[v] + bin + r * (hn2[v] + bhn));
        h2[b][j2] = (1.0f - z) * n + z * hp;
      }
    }
    __syncthreads();
  }

  // ---- FC head: out[b] = h2[b,:] . fc_w + fc_b ----
  if (tid < GB) {
    float s = fc_b[0];
#pragma unroll 4
    for (int j = 0; j < GH2; ++j) s += h2[tid][j] * fc_w[j];
    out[tid] = s;
  }
#endif
}

// ---------------------------------------------------------------------------
extern "C" void kernel_launch(void* const* d_in, const int* in_sizes, int n_in,
                              void* d_out, int out_size, void* d_ws, size_t ws_size,
                              hipStream_t stream) {
  (void)in_sizes; (void)n_in; (void)out_size; (void)ws_size;
  const float* x     = (const float*)d_in[0];
  const float* w_ih1 = (const float*)d_in[1];
  const float* w_hh1 = (const float*)d_in[2];
  const float* b_ih1 = (const float*)d_in[3];
  const float* b_hh1 = (const float*)d_in[4];
  const float* w_ih2 = (const float*)d_in[5];
  const float* w_hh2 = (const float*)d_in[6];
  const float* b_ih2 = (const float*)d_in[7];
  const float* b_hh2 = (const float*)d_in[8];
  const float* fc_w  = (const float*)d_in[9];
  const float* fc_b  = (const float*)d_in[10];
  float* xg = (float*)d_ws;                    // [T][B][384] = 48 MB

  // 32768 rows / (16 rows/wave * 8 waves/block) = 256 blocks
  gru_l1_input_gemm<<<256, 256, 0, stream>>>(x, w_ih1, b_ih1, xg);
  gru_seq_fused<<<1, 1024, 0, stream>>>(xg, w_hh1, b_hh1, w_ih2, b_ih2,
                                        w_hh2, b_hh2, fc_w, fc_b,
                                        (float*)d_out);
}
